// Block_24180665876821
// MI455X (gfx1250) — compile-verified, matrix operand order checked by reading, not run on hardware
//
#include <hip/hip_runtime.h>
#include <hip/hip_bf16.h>
#include <cmath>

#define DIMC   512
#define HEADS  8
#define HEADD  64
#define HIDDEN 2048
#define BATCH  2
#define HH     56
#define WWID   56
#define NTOK   3136   /* 56*56 */
#define ROWS   6272   /* BATCH*NTOK */
#define EPSLN  1e-5f

typedef __attribute__((ext_vector_type(16))) _Float16 v16h;
typedef __attribute__((ext_vector_type(8)))  _Float16 v8h;
typedef __attribute__((ext_vector_type(8)))  float    v8f;

union FragU { v16h v; v8h h[2]; };

// ---------------------------------------------------------------------------
// Fragment loader for 16-bit WMMA A (16x32 MxK) and B (32x16 KxN, loaded from
// the N-major/K-contiguous transpose) per CDNA5 ISA 7.12.2:
//   lane&15 = M (or N); halves 0..7 -> K = hi8+0..7, halves 8..15 -> K = 16+hi8+0..7
//   where hi8 = 8*(lane>=16).  Two contiguous b128 loads per lane.
// ---------------------------------------------------------------------------
__device__ __forceinline__ v16h load_frag(const _Float16* __restrict__ p,
                                          int row0, int k0, int ld) {
    int lane = threadIdx.x & 31;
    int r    = lane & 15;
    int hi   = (lane >> 4) << 3;
    const _Float16* base = p + (size_t)(row0 + r) * (size_t)ld + k0 + hi;
    FragU f;
    f.h[0] = *(const v8h*)(base);
    f.h[1] = *(const v8h*)(base + 16);
    return f.v;
}

__device__ __forceinline__ v8f wmma_f16(v16h a, v16h b, v8f c) {
    return __builtin_amdgcn_wmma_f32_16x16x32_f16(
        /*neg_a=*/false, a, /*neg_b=*/false, b,
        /*c_mod=*/(short)0, c, /*reuse_a=*/false, /*reuse_b=*/false);
}

__device__ __forceinline__ v8f zero8() {
    v8f z;
#pragma unroll
    for (int i = 0; i < 8; ++i) z[i] = 0.0f;
    return z;
}

// ---------------------------------------------------------------------------
// DPP16-based reductions across the 16-lane rows of the WMMA C-layout.
// xor1 = quad_perm(1,0,3,2)=0xB1, xor2 = quad_perm(2,3,0,1)=0x4E,
// xor4 = row_half_mirror=0x141,   xor8 = row_mirror=0x140.
// Pure VALU; no ds_bpermute traffic in the softmax hot loop.
// dpp_ctrl must be a literal constant -> template parameter.
// ---------------------------------------------------------------------------
template<int CTRL>
__device__ __forceinline__ float dpp_movf(float x) {
    int xi = __builtin_bit_cast(int, x);
    int r  = __builtin_amdgcn_update_dpp(xi, xi, CTRL, 0xF, 0xF, true);
    return __builtin_bit_cast(float, r);
}
__device__ __forceinline__ float red16_max(float x) {
    x = fmaxf(x, dpp_movf<0xB1>(x));
    x = fmaxf(x, dpp_movf<0x4E>(x));
    x = fmaxf(x, dpp_movf<0x141>(x));
    x = fmaxf(x, dpp_movf<0x140>(x));
    return x;
}
__device__ __forceinline__ float red16_sum(float x) {
    x += dpp_movf<0xB1>(x);
    x += dpp_movf<0x4E>(x);
    x += dpp_movf<0x141>(x);
    x += dpp_movf<0x140>(x);
    return x;
}

// ---------------------------------------------------------------------------
// fp32 -> fp16 weight/activation conversion
// ---------------------------------------------------------------------------
__global__ void __launch_bounds__(256) f32_to_f16_kernel(
    const float* __restrict__ src, _Float16* __restrict__ dst, int n) {
    int i = blockIdx.x * 256 + threadIdx.x;
    if (i < n) dst[i] = (_Float16)src[i];
}

// ---------------------------------------------------------------------------
// LayerNorm over DIMC=512 per row; one wave per row; fp32 in -> f16 out
// ---------------------------------------------------------------------------
__global__ void __launch_bounds__(256) ln_kernel(
    const float* __restrict__ src, const float* __restrict__ w,
    const float* __restrict__ b, _Float16* __restrict__ dst, int rows) {
    int wid  = threadIdx.x >> 5;
    int lane = threadIdx.x & 31;
    int row  = blockIdx.x * 8 + wid;
    if (row >= rows) return;
    const float* p = src + (size_t)row * DIMC;
    float vals[16];
    float s = 0.0f, s2 = 0.0f;
#pragma unroll
    for (int i = 0; i < 16; ++i) {
        float v = p[i * 32 + lane];
        vals[i] = v; s += v; s2 += v * v;
    }
#pragma unroll
    for (int off = 16; off >= 1; off >>= 1) {
        s  += __shfl_xor(s,  off, 32);
        s2 += __shfl_xor(s2, off, 32);
    }
    float mean = s * (1.0f / DIMC);
    float var  = s2 * (1.0f / DIMC) - mean * mean;
    float rstd = rsqrtf(var + EPSLN);
    _Float16* o = dst + (size_t)row * DIMC;
#pragma unroll
    for (int i = 0; i < 16; ++i) {
        int c = i * 32 + lane;
        o[c] = (_Float16)((vals[i] - mean) * rstd * w[c] + b[c]);
    }
}

// ---------------------------------------------------------------------------
// WMMA GEMM:  out[M x N] = A[M x K] * Wt[N x K]^T + bias (+ res)
// A row-major f16 (lda), Wt row-major f16 (ldw).  Wave tile 64x64 (4x4 WMMA),
// block = 8 waves -> 128x256.  M % 128 == 0, N % 256 == 0, K % 64 == 0.
// ---------------------------------------------------------------------------
template<bool OUT_F16, bool RES>
__global__ void __launch_bounds__(256) gemm_wmma_kernel(
    const _Float16* __restrict__ A, int lda,
    const _Float16* __restrict__ Wt, int ldw,
    const float* __restrict__ bias,
    const float* __restrict__ res, int ldres,
    void* __restrict__ out, int ldo, int K) {
    int wid  = threadIdx.x >> 5;
    int lane = threadIdx.x & 31;
    int m0 = blockIdx.x * 128 + (wid & 1) * 64;
    int n0 = blockIdx.y * 256 + (wid >> 1) * 64;

    v8f acc[4][4];
#pragma unroll
    for (int i = 0; i < 4; ++i)
#pragma unroll
        for (int j = 0; j < 4; ++j) acc[i][j] = zero8();

#pragma unroll 2
    for (int k = 0; k < K; k += 32) {
        v16h af[4], bf[4];
#pragma unroll
        for (int i = 0; i < 4; ++i) af[i] = load_frag(A,  m0 + 16 * i, k, lda);
#pragma unroll
        for (int j = 0; j < 4; ++j) bf[j] = load_frag(Wt, n0 + 16 * j, k, ldw);
#pragma unroll
        for (int i = 0; i < 4; ++i)
#pragma unroll
            for (int j = 0; j < 4; ++j)
                acc[i][j] = wmma_f16(af[i], bf[j], acc[i][j]);
    }

    // C/D layout: VGPR r, lanes 0-15 -> row r, col lane; lanes 16-31 -> row r+8
    int col   = lane & 15;
    int rbase = (lane >> 4) << 3;
#pragma unroll
    for (int j = 0; j < 4; ++j) {
        int cg = n0 + j * 16 + col;
        float bv = bias[cg];
#pragma unroll
        for (int i = 0; i < 4; ++i) {
#pragma unroll
            for (int r = 0; r < 8; ++r) {
                int rg = m0 + i * 16 + rbase + r;
                float v = acc[i][j][r] + bv;
                if (RES) v += res[(size_t)rg * ldres + cg];
                if (OUT_F16)
                    ((_Float16*)out)[(size_t)rg * ldo + cg] = (_Float16)v;
                else
                    ((float*)out)[(size_t)rg * ldo + cg] = v;
            }
        }
    }
}

// ---------------------------------------------------------------------------
// Flash attention: grid (N/64, B*HEADS), block 128 (4 waves).
// Each wave owns 16 query rows; keys processed in blocks of 64
// (16 WMMAs per wave per barrier pair).
// qkv f16 [B][N][1536]  (q|k|v, each [HEADS][64]);  out f16 [B][N][512].
// ---------------------------------------------------------------------------
__global__ void __launch_bounds__(128) attn_kernel(
    const _Float16* __restrict__ qkv, const float* __restrict__ temp_p,
    _Float16* __restrict__ out) {
    __shared__ __attribute__((aligned(16))) _Float16 Vt[64][72];      // V^T tile: [d][key]
    __shared__ __attribute__((aligned(16))) _Float16 Pb[4][16][72];   // per-wave P tile

    int bh = blockIdx.y;
    int b  = bh >> 3;
    int h  = bh & 7;
    int wid  = threadIdx.x >> 5;
    int lane = threadIdx.x & 31;
    int q0 = blockIdx.x * 64 + wid * 16;

    const _Float16* qp = qkv + (size_t)b * NTOK * 1536 + h * HEADD;
    const _Float16* kp = qp + DIMC;
    const _Float16* vp = qp + 2 * DIMC;
    float inv_t = 1.0f / temp_p[0];

    v16h aq0 = load_frag(qp, q0, 0, 1536);
    v16h aq1 = load_frag(qp, q0, 32, 1536);

    float mrow[8], lrow[8];
    v8f oa[4];
#pragma unroll
    for (int r = 0; r < 8; ++r) { mrow[r] = -1e30f; lrow[r] = 0.0f; }
#pragma unroll
    for (int j = 0; j < 4; ++j) oa[j] = zero8();

    int col   = lane & 15;
    int rbase = (lane >> 4) << 3;

    for (int kb = 0; kb < NTOK; kb += 64) {
        // cooperative transpose-stage of V block [64 keys][64 d] -> Vt[d][key]
        {
            int key = threadIdx.x >> 1;          // 0..63
            int d0  = (threadIdx.x & 1) * 32;    // 0 or 32
            const _Float16* src = vp + (size_t)(kb + key) * 1536 + d0;
#pragma unroll
            for (int e = 0; e < 32; ++e) Vt[d0 + e][key] = src[e];
        }
        __syncthreads();

        // S = Q K^T for 16 queries x 64 keys (8 WMMAs)
        v8f s[4];
#pragma unroll
        for (int nb = 0; nb < 4; ++nb) {
            s[nb] = zero8();
            s[nb] = wmma_f16(aq0, load_frag(kp, kb + 16 * nb, 0,  1536), s[nb]);
            s[nb] = wmma_f16(aq1, load_frag(kp, kb + 16 * nb, 32, 1536), s[nb]);
        }

        // online softmax: row reductions via DPP16 (no LDS traffic)
        float scale[8];
#pragma unroll
        for (int r = 0; r < 8; ++r) {
            float v0 = s[0][r] * inv_t;
            float v1 = s[1][r] * inv_t;
            float v2 = s[2][r] * inv_t;
            float v3 = s[3][r] * inv_t;
            float mx = red16_max(fmaxf(fmaxf(v0, v1), fmaxf(v2, v3)));
            float mn = fmaxf(mrow[r], mx);
            scale[r] = __expf(mrow[r] - mn);
            float p0 = __expf(v0 - mn);
            float p1 = __expf(v1 - mn);
            float p2 = __expf(v2 - mn);
            float p3 = __expf(v3 - mn);
            s[0][r] = p0; s[1][r] = p1; s[2][r] = p2; s[3][r] = p3;
            float rs = red16_sum((p0 + p1) + (p2 + p3));
            lrow[r] = lrow[r] * scale[r] + rs;
            mrow[r] = mn;
        }
#pragma unroll
        for (int j = 0; j < 4; ++j)
#pragma unroll
            for (int r = 0; r < 8; ++r) oa[j][r] *= scale[r];

        // C-layout -> A-fragment reshape of P through LDS
#pragma unroll
        for (int r = 0; r < 8; ++r) {
#pragma unroll
            for (int i = 0; i < 4; ++i)
                Pb[wid][rbase + r][i * 16 + col] = (_Float16)s[i][r];
        }
        v16h ap0 = load_frag(&Pb[wid][0][0], 0, 0,  72);
        v16h ap1 = load_frag(&Pb[wid][0][0], 0, 32, 72);
#pragma unroll
        for (int j = 0; j < 4; ++j) {
            oa[j] = wmma_f16(ap0, load_frag(&Vt[0][0], j * 16, 0,  72), oa[j]);
            oa[j] = wmma_f16(ap1, load_frag(&Vt[0][0], j * 16, 32, 72), oa[j]);
        }
        __syncthreads();
    }

    _Float16* op = out + ((size_t)b * NTOK + q0) * DIMC + h * HEADD;
#pragma unroll
    for (int r = 0; r < 8; ++r) {
        float inv_l = 1.0f / lrow[r];
#pragma unroll
        for (int j = 0; j < 4; ++j)
            op[(size_t)(rbase + r) * DIMC + j * 16 + col] =
                (_Float16)(oa[j][r] * inv_l);
    }
}

// ---------------------------------------------------------------------------
// Depthwise 3x3 conv (SAME) + bias + exact GELU.
// h1/g layout: [B][N][HIDDEN] f16 (channel contiguous for coalescing).
// ---------------------------------------------------------------------------
__global__ void __launch_bounds__(256) dwconv_gelu_kernel(
    const _Float16* __restrict__ h1, const float* __restrict__ w,
    const float* __restrict__ bias, _Float16* __restrict__ g) {
    size_t idx = (size_t)blockIdx.x * 256 + threadIdx.x;
    if (idx >= (size_t)ROWS * HIDDEN) return;
    int c  = (int)(idx & (HIDDEN - 1));
    size_t nb = idx >> 11;
    int n = (int)(nb % NTOK);
    int b = (int)(nb / NTOK);
    int y = n / WWID, x = n % WWID;
    const float* wc = w + c * 9;
    float acc = bias[c];
#pragma unroll
    for (int dy = -1; dy <= 1; ++dy) {
        int yy = y + dy;
        if (yy < 0 || yy >= HH) continue;
#pragma unroll
        for (int dx = -1; dx <= 1; ++dx) {
            int xx = x + dx;
            if (xx < 0 || xx >= WWID) continue;
            float v = (float)h1[((size_t)b * NTOK + yy * WWID + xx) * HIDDEN + c];
            acc += v * wc[(dy + 1) * 3 + (dx + 1)];
        }
    }
    float gv = 0.5f * acc * (1.0f + erff(acc * 0.70710678118f));
    g[idx] = (_Float16)gv;
}

// ---------------------------------------------------------------------------
// host-side orchestration
// ---------------------------------------------------------------------------
extern "C" void kernel_launch(void* const* d_in, const int* in_sizes, int n_in,
                              void* d_out, int out_size, void* d_ws, size_t ws_size,
                              hipStream_t stream) {
    (void)in_sizes; (void)n_in; (void)out_size; (void)ws_size;
    const float* x      = (const float*)d_in[0];
    const float* ln1_w  = (const float*)d_in[1];
    const float* ln1_b  = (const float*)d_in[2];
    const float* qkv_w  = (const float*)d_in[3];
    const float* qkv_b  = (const float*)d_in[4];
    const float* out_w  = (const float*)d_in[5];
    const float* out_b  = (const float*)d_in[6];
    const float* temp   = (const float*)d_in[7];
    const float* ln2_w  = (const float*)d_in[8];
    const float* ln2_b  = (const float*)d_in[9];
    const float* fc1_w  = (const float*)d_in[10];
    const float* fc1_b  = (const float*)d_in[11];
    const float* dw_w   = (const float*)d_in[12];
    const float* dw_b   = (const float*)d_in[13];
    const float* fc2_w  = (const float*)d_in[14];
    const float* fc2_b  = (const float*)d_in[15];
    float* outp = (float*)d_out;

    // workspace carving (f16 buffers, 256B aligned)
    char* ws = (char*)d_ws;
    size_t off = 0;
    auto carve = [&](size_t bytes) {
        char* p = ws + off;
        off += (bytes + 255) & ~(size_t)255;
        return p;
    };
    _Float16* wq   = (_Float16*)carve((size_t)3 * DIMC * DIMC * 2);      // 1536x512
    _Float16* wo   = (_Float16*)carve((size_t)DIMC * DIMC * 2);          // 512x512
    _Float16* w1   = (_Float16*)carve((size_t)HIDDEN * DIMC * 2);        // 2048x512
    _Float16* w2   = (_Float16*)carve((size_t)DIMC * HIDDEN * 2);        // 512x2048
    _Float16* xn   = (_Float16*)carve((size_t)ROWS * DIMC * 2);          // LN output (reused)
    _Float16* qkvb = (_Float16*)carve((size_t)ROWS * 3 * DIMC * 2);      // QKV
    _Float16* ao   = (_Float16*)carve((size_t)ROWS * DIMC * 2);          // attention output
    float*    y1   = (float*)   carve((size_t)ROWS * DIMC * 4);          // first residual sum
    _Float16* h1   = (_Float16*)carve((size_t)ROWS * HIDDEN * 2);        // fc1 output
    _Float16* gb   = (_Float16*)carve((size_t)ROWS * HIDDEN * 2);        // conv+gelu output

    // weight conversions fp32 -> fp16
    {
        int n;
        n = 3 * DIMC * DIMC;
        f32_to_f16_kernel<<<(n + 255) / 256, 256, 0, stream>>>(qkv_w, wq, n);
        n = DIMC * DIMC;
        f32_to_f16_kernel<<<(n + 255) / 256, 256, 0, stream>>>(out_w, wo, n);
        n = HIDDEN * DIMC;
        f32_to_f16_kernel<<<(n + 255) / 256, 256, 0, stream>>>(fc1_w, w1, n);
        n = DIMC * HIDDEN;
        f32_to_f16_kernel<<<(n + 255) / 256, 256, 0, stream>>>(fc2_w, w2, n);
    }

    // 1) LN1
    ln_kernel<<<ROWS / 8, 256, 0, stream>>>(x, ln1_w, ln1_b, xn, ROWS);

    // 2) QKV: [6272,512] x [1536,512]^T -> f16 [6272,1536]
    gemm_wmma_kernel<true, false><<<dim3(ROWS / 128, (3 * DIMC) / 256), 256, 0, stream>>>(
        xn, DIMC, wq, DIMC, qkv_b, nullptr, 0, qkvb, 3 * DIMC, DIMC);

    // 3) flash attention -> f16 [6272,512]
    attn_kernel<<<dim3(NTOK / 64, BATCH * HEADS), 128, 0, stream>>>(qkvb, temp, ao);

    // 4) out-proj + residual(x): -> f32 y1 [6272,512]
    gemm_wmma_kernel<false, true><<<dim3(ROWS / 128, DIMC / 256), 256, 0, stream>>>(
        ao, DIMC, wo, DIMC, out_b, x, DIMC, y1, DIMC, DIMC);

    // 5) LN2
    ln_kernel<<<ROWS / 8, 256, 0, stream>>>(y1, ln2_w, ln2_b, xn, ROWS);

    // 6) fc1: -> f16 [6272,2048]
    gemm_wmma_kernel<true, false><<<dim3(ROWS / 128, HIDDEN / 256), 256, 0, stream>>>(
        xn, DIMC, w1, DIMC, fc1_b, nullptr, 0, h1, HIDDEN, DIMC);

    // 7) depthwise conv 3x3 + GELU -> f16 [6272,2048]
    {
        size_t total = (size_t)ROWS * HIDDEN;
        dwconv_gelu_kernel<<<(unsigned)((total + 255) / 256), 256, 0, stream>>>(
            h1, dw_w, dw_b, gb);
    }

    // 8) fc2 + residual(y1): -> f32 d_out [6272,512]
    gemm_wmma_kernel<false, true><<<dim3(ROWS / 128, DIMC / 256), 256, 0, stream>>>(
        gb, HIDDEN, w2, HIDDEN, fc2_b, y1, DIMC, outp, DIMC, HIDDEN);
}
